// PytorchTransformer_58445914964305
// MI455X (gfx1250) — matrix operand order repaired and checked by reading, not compiled, not run on hardware
//
#include <hip/hip_runtime.h>
#include <hip/hip_bf16.h>
#include <math.h>
#include <stdint.h>

// ---- model constants (match reference) ----
#define CC   768          // n_embd
#define HH   8            // n_head
#define HSZ  96           // head_size
#define EE   8            // n_experts
#define LL   12           // n_layer
#define VV   95           // vocab
#define TT   64           // seq len (== block_size)
#define BB   16           // batch
#define NTOK (BB*TT)      // 1024 tokens
#define FFD  (4*CC)       // 3072

typedef __bf16 bf16_t;
typedef __attribute__((ext_vector_type(16))) __bf16 v16bf;
typedef __attribute__((ext_vector_type(8)))  __bf16 v8bf;
typedef __attribute__((ext_vector_type(4)))  __bf16 v4bf;
typedef __attribute__((ext_vector_type(8)))  float  v8f;

// ================= WMMA GEMM =================
// Out[M x Nn] = epilogue( A[M x K] @ Bw[K x Nn] + bias )
// ACT: 0 = none, 1 = exact GELU
// EPI: 0 = store, 1 = Out += rowscale[row]*val (MoE combine)
// Block 64x64, BK=32, 4 waves (each 32x32 = 2x2 WMMA tiles),
// double-buffered LDS, B stored transposed so fragments are 2x ds_load_b128.
#define BMT 64
#define BNT 64
#define BKT 32
#define LDP 40   // padded row length (80B: 16B-aligned, odd bank stride)

__device__ __forceinline__ float gelu_exact(float x) {
  return 0.5f * x * (1.0f + erff(x * 0.70710678118654752440f));
}

template<int ACT, int EPI>
__global__ __launch_bounds__(128)
void gemm_bf16_wmma(const float* __restrict__ A, int lda,
                    const float* __restrict__ Bw, int ldb,
                    const float* __restrict__ bias,
                    float* __restrict__ Out, int ldo,
                    int M, int Nn, int K,
                    const float* __restrict__ rowscale, int sstride, int soff)
{
  __shared__ __align__(16) bf16_t As [2][BMT][LDP];   // row-major A tile
  __shared__ __align__(16) bf16_t BsT[2][BNT][LDP];   // transposed B tile

  const int bm     = blockIdx.y * BMT;
  const int bn     = blockIdx.x * BNT;
  const int tid    = threadIdx.x;
  const int lane   = tid & 31;
  const int wave   = tid >> 5;
  const int laneLo = lane & 15;
  const int laneHi = lane >> 4;
  const int wm     = (wave >> 1) * 32;   // wave quadrant
  const int wn     = (wave & 1)  * 32;

  // staging assignments
  const int ar [4] = { (tid + 0) >> 3, (tid + 128) >> 3, (tid + 256) >> 3, (tid + 384) >> 3 };
  const int ac4    = (tid & 7) << 2;
  const int colB   = tid & 63;           // B: thread owns one column, 16 rows
  const int rbase  = (tid >> 6) * 16;    // 0 or 16
  const bool colOK = (bn + colB) < Nn;

  float4 fa[4];
  float  tb[16];

  // issue all global loads for K-tile k0 into registers
  auto load_global = [&](int k0) {
#pragma unroll
    for (int i = 0; i < 4; i++)
      fa[i] = *(const float4*)(A + (size_t)(bm + ar[i]) * lda + k0 + ac4);
#pragma unroll
    for (int i = 0; i < 16; i++)
      tb[i] = colOK ? Bw[(size_t)(k0 + rbase + i) * ldb + bn + colB] : 0.0f;
  };

  // convert + store registers into LDS buffer `buf`
  auto store_lds = [&](int buf) {
#pragma unroll
    for (int i = 0; i < 4; i++) {
      v4bf p; p[0] = (bf16_t)fa[i].x; p[1] = (bf16_t)fa[i].y;
              p[2] = (bf16_t)fa[i].z; p[3] = (bf16_t)fa[i].w;
      *(v4bf*)&As[buf][ar[i]][ac4] = p;
    }
    v8bf blo, bhi;
#pragma unroll
    for (int i = 0; i < 8; i++) { blo[i] = (bf16_t)tb[i]; bhi[i] = (bf16_t)tb[8 + i]; }
    *(v8bf*)&BsT[buf][colB][rbase + 0] = blo;
    *(v8bf*)&BsT[buf][colB][rbase + 8] = bhi;
  };

  v8f acc[2][2] = {};

  load_global(0);
  store_lds(0);
  __syncthreads();

  const int nk = K / BKT;
  for (int t = 0; t < nk; t++) {
    const int cur  = t & 1;
    const bool more = (t + 1 < nk);
    if (more) load_global((t + 1) * BKT);          // pipeline next tile
    if (t + 2 < nk) {                              // warm L2 for t+2
      __builtin_prefetch(A  + (size_t)(bm + (tid >> 1)) * lda + (t + 2) * BKT, 0, 1);
      __builtin_prefetch(Bw + (size_t)((t + 2) * BKT + (tid >> 2)) * ldb + bn, 0, 1);
    }

    // --- fragments: two contiguous 16B chunks each (ds_load_b128 x2) ---
    v16bf af[2], bfv[2];
#pragma unroll
    for (int tm = 0; tm < 2; tm++) {
      const int row = wm + tm * 16 + laneLo;
      v8bf lo = *(const v8bf*)&As[cur][row][8 * laneHi];        // K = 8*hi .. +7
      v8bf hi = *(const v8bf*)&As[cur][row][16 + 8 * laneHi];   // K = 16+8*hi .. +7
      af[tm] = __builtin_shufflevector(lo, hi, 0,1,2,3,4,5,6,7,8,9,10,11,12,13,14,15);
    }
#pragma unroll
    for (int tn = 0; tn < 2; tn++) {
      const int col = wn + tn * 16 + laneLo;
      v8bf lo = *(const v8bf*)&BsT[cur][col][8 * laneHi];       // K = 8*hi .. +7
      v8bf hi = *(const v8bf*)&BsT[cur][col][16 + 8 * laneHi];  // K = 16+8*hi .. +7
      bfv[tn] = __builtin_shufflevector(lo, hi, 0,1,2,3,4,5,6,7,8,9,10,11,12,13,14,15);
    }
#pragma unroll
    for (int tm = 0; tm < 2; tm++)
#pragma unroll
      for (int tn = 0; tn < 2; tn++)
        acc[tm][tn] = __builtin_amdgcn_wmma_f32_16x16x32_bf16(
            false, af[tm], false, bfv[tn], (short)0, acc[tm][tn], false, false);

    if (more) store_lds(cur ^ 1);                  // fill other buffer
    __syncthreads();                               // one barrier per K-step
  }

  // --- epilogue (C layout: vgpr r -> M = r + 8*laneHi, N = laneLo) ---
#pragma unroll
  for (int tm = 0; tm < 2; tm++) {
#pragma unroll
    for (int tn = 0; tn < 2; tn++) {
      const int col = bn + wn + tn * 16 + laneLo;
      if (col < Nn) {
        const float bv = bias ? bias[col] : 0.0f;
#pragma unroll
        for (int r = 0; r < 8; r++) {
          const int row = bm + wm + tm * 16 + r + 8 * laneHi;
          float v = acc[tm][tn][r] + bv;
          if (ACT == 1) v = gelu_exact(v);
          if (EPI == 0) {
            Out[(size_t)row * ldo + col] = v;
          } else {
            const float s = rowscale[(size_t)row * sstride + soff];
            if (s != 0.0f) Out[(size_t)row * ldo + col] += s * v;
          }
        }
      }
    }
  }
}

// ================= embedding =================
__global__ __launch_bounds__(256)
void embed_kernel(const int* __restrict__ idx, const float* __restrict__ tok,
                  const float* __restrict__ pos, float* __restrict__ x)
{
  int gid = blockIdx.x * 256 + threadIdx.x;   // < NTOK*CC
  int n = gid / CC, c = gid % CC;
  int t = n % TT;                              // t % BS == t (BS==T)
  x[gid] = tok[(size_t)idx[n] * CC + c] + pos[(size_t)t * CC + c];
}

__global__ __launch_bounds__(256)
void zero_kernel(float* __restrict__ p, int n)
{
  int g = blockIdx.x * 256 + threadIdx.x;
  if (g < n) p[g] = 0.0f;
}

// ================= attention (per (b,h) block) =================
// K/V tiles staged via gfx1250 async global->LDS DMA (ASYNCcnt path).
__global__ __launch_bounds__(256)
void attn_kernel(const float* __restrict__ q, const float* __restrict__ k,
                 const float* __restrict__ v, float* __restrict__ o)
{
  __shared__ __align__(16) float Ks[TT][HSZ];
  __shared__ __align__(16) float Vs[TT][HSZ];
  __shared__ float qs[8][HSZ];
  __shared__ float wsh[8][TT];

  const int bh = blockIdx.x;
  const int b  = bh / HH;
  const int hh = bh % HH;
  const int tid  = threadIdx.x;
  const int wave = tid >> 5;
  const int lane = tid & 31;

  const float* kbase = k + (size_t)(b * TT) * CC + hh * HSZ;
  const float* vbase = v + (size_t)(b * TT) * CC + hh * HSZ;

  // 64x96 f32 per matrix = 1536 x 16B granules; 256 thr x 6 each.
  // global_load_async_to_lds_b128: vdst = per-lane LDS byte addr (low 32 bits
  // of the generic address), vaddr = per-lane 64-bit global addr.
#pragma unroll
  for (int i = 0; i < 6; i++) {
    int id4 = tid + 256 * i;        // 0..1535
    int s   = id4 / 24;             // row 0..63
    int d   = (id4 % 24) * 4;       // col 0..92 (16B aligned)
    uint32_t ldsK = (uint32_t)(uintptr_t)&Ks[s][d];
    uint64_t gaK  = (uint64_t)(uintptr_t)(kbase + (size_t)s * CC + d);
    asm volatile("global_load_async_to_lds_b128 %0, %1, off"
                 :: "v"(ldsK), "v"(gaK) : "memory");
    uint32_t ldsV = (uint32_t)(uintptr_t)&Vs[s][d];
    uint64_t gaV  = (uint64_t)(uintptr_t)(vbase + (size_t)s * CC + d);
    asm volatile("global_load_async_to_lds_b128 %0, %1, off"
                 :: "v"(ldsV), "v"(gaV) : "memory");
  }
  asm volatile("s_wait_asynccnt 0x0" ::: "memory");
  __syncthreads();

  const float scl = 0.10206207261596575f;  // 96^-0.5
  for (int i = 0; i < 8; i++) {
    int t = wave * 8 + i;
#pragma unroll
    for (int j = 0; j < 3; j++)
      qs[wave][lane * 3 + j] = q[(size_t)(b * TT + t) * CC + hh * HSZ + lane * 3 + j];
    __syncthreads();

    int s0 = lane, s1 = lane + 32;
    float sc0 = -3.402823466e38f, sc1 = -3.402823466e38f;
    if (s0 <= t) { float d0 = 0.f; for (int d = 0; d < HSZ; d++) d0 += qs[wave][d] * Ks[s0][d]; sc0 = d0 * scl; }
    if (s1 <= t) { float d1 = 0.f; for (int d = 0; d < HSZ; d++) d1 += qs[wave][d] * Ks[s1][d]; sc1 = d1 * scl; }
    float m = fmaxf(sc0, sc1);
    for (int off = 16; off; off >>= 1) m = fmaxf(m, __shfl_xor(m, off));
    float e0 = (s0 <= t) ? expf(sc0 - m) : 0.0f;
    float e1 = (s1 <= t) ? expf(sc1 - m) : 0.0f;
    float sum = e0 + e1;
    for (int off = 16; off; off >>= 1) sum += __shfl_xor(sum, off);
    float inv = 1.0f / sum;
    wsh[wave][s0] = e0 * inv;
    wsh[wave][s1] = e1 * inv;
    __syncthreads();

#pragma unroll
    for (int j = 0; j < 3; j++) {
      int d = lane * 3 + j;
      float a = 0.0f;
      for (int s = 0; s <= t; s++) a += wsh[wave][s] * Vs[s][d];
      o[(size_t)(b * TT + t) * CC + hh * HSZ + d] = a;
    }
    __syncthreads();
  }
}

// ================= residual + layernorm =================
template<bool RESID>
__global__ __launch_bounds__(256)
void ln_kernel(const float* __restrict__ xin, const float* __restrict__ yin,
               const float* __restrict__ g, const float* __restrict__ bta,
               float* __restrict__ xout)
{
  __shared__ float red[8];
  const int row = blockIdx.x;
  const int tid = threadIdx.x;
  float vloc[3];
  float s = 0.0f;
#pragma unroll
  for (int j = 0; j < 3; j++) {
    int c = tid + 256 * j;
    float t = xin[(size_t)row * CC + c];
    if (RESID) t += yin[(size_t)row * CC + c];
    vloc[j] = t;
    s += t;
  }
  for (int off = 16; off; off >>= 1) s += __shfl_xor(s, off);
  if ((tid & 31) == 0) red[tid >> 5] = s;
  __syncthreads();
  float tot = 0.0f;
#pragma unroll
  for (int w = 0; w < 8; w++) tot += red[w];
  const float mu = tot * (1.0f / CC);
  __syncthreads();
  float ss = 0.0f;
#pragma unroll
  for (int j = 0; j < 3; j++) { float d = vloc[j] - mu; ss += d * d; }
  for (int off = 16; off; off >>= 1) ss += __shfl_xor(ss, off);
  if ((tid & 31) == 0) red[tid >> 5] = ss;
  __syncthreads();
  float tot2 = 0.0f;
#pragma unroll
  for (int w = 0; w < 8; w++) tot2 += red[w];
  const float inv = rsqrtf(tot2 * (1.0f / CC) + 1e-5f);
#pragma unroll
  for (int j = 0; j < 3; j++) {
    int c = tid + 256 * j;
    xout[(size_t)row * CC + c] = (vloc[j] - mu) * inv * g[c] + bta[c];
  }
}

// ================= gating + top-2 routing weights =================
__global__ __launch_bounds__(32)
void gate_topk_kernel(const float* __restrict__ x, const float* __restrict__ gw,
                      const float* __restrict__ gb, float* __restrict__ route)
{
  const int n = blockIdx.x;
  const int lane = threadIdx.x;
  float logit = -3.402823466e38f;
  if (lane < EE) {
    float a = gb[lane];
    for (int c = 0; c < CC; c++) a += x[(size_t)n * CC + c] * gw[(size_t)c * EE + lane];
    logit = a;
  }
  float m = logit;
  for (int off = 4; off; off >>= 1) m = fmaxf(m, __shfl_xor(m, off));
  float ex = (lane < EE) ? expf(logit - m) : 0.0f;
  float sum = ex;
  for (int off = 4; off; off >>= 1) sum += __shfl_xor(sum, off);
  float p = ex / sum;
  float pj[EE];
#pragma unroll
  for (int j = 0; j < EE; j++) pj[j] = __shfl(p, j);
  int i1 = 0; float b1v = pj[0];
#pragma unroll
  for (int j = 1; j < EE; j++) if (pj[j] > b1v) { b1v = pj[j]; i1 = j; }
  int i2 = -1; float b2v = -1.0f;
#pragma unroll
  for (int j = 0; j < EE; j++) if (j != i1 && pj[j] > b2v) { b2v = pj[j]; i2 = j; }
  const float wsum = b1v + b2v;
  if (lane < EE)
    route[(size_t)n * EE + lane] = (lane == i1) ? (b1v / wsum)
                                 : (lane == i2) ? (b2v / wsum) : 0.0f;
}

// ================= host orchestration =================
extern "C" void kernel_launch(void* const* d_in, const int* in_sizes, int n_in,
                              void* d_out, int out_size, void* d_ws, size_t ws_size,
                              hipStream_t stream) {
  (void)in_sizes; (void)n_in; (void)out_size; (void)ws_size;
  const int*   idx     = (const int*)  d_in[0];
  const float* tok_emb = (const float*)d_in[1];
  const float* pos_emb = (const float*)d_in[2];
  const float* Wq      = (const float*)d_in[3];
  const float* Wk      = (const float*)d_in[4];
  const float* Wv      = (const float*)d_in[5];
  const float* Wo      = (const float*)d_in[6];
  const float* bo      = (const float*)d_in[7];
  const float* gate_W  = (const float*)d_in[8];
  const float* gate_b  = (const float*)d_in[9];
  const float* W1      = (const float*)d_in[10];
  const float* b1      = (const float*)d_in[11];
  const float* W2      = (const float*)d_in[12];
  const float* b2      = (const float*)d_in[13];
  const float* ln1_g   = (const float*)d_in[14];
  const float* ln1_b   = (const float*)d_in[15];
  const float* ln2_g   = (const float*)d_in[16];
  const float* ln2_b   = (const float*)d_in[17];
  const float* lnf_g   = (const float*)d_in[18];
  const float* lnf_b   = (const float*)d_in[19];
  const float* head_W  = (const float*)d_in[20];
  const float* head_b  = (const float*)d_in[21];
  float* out = (float*)d_out;

  // workspace carve-up (floats)
  float* ws = (float*)d_ws;
  size_t ofs = 0;
  float* x     = ws + ofs; ofs += (size_t)NTOK * CC;
  float* q     = ws + ofs; ofs += (size_t)NTOK * CC;
  float* kbuf  = ws + ofs; ofs += (size_t)NTOK * CC;
  float* vbuf  = ws + ofs; ofs += (size_t)NTOK * CC;
  float* obuf  = ws + ofs; ofs += (size_t)NTOK * CC;
  float* y     = ws + ofs; ofs += (size_t)NTOK * CC;
  float* hbuf  = ws + ofs; ofs += (size_t)NTOK * FFD;
  float* route = ws + ofs; ofs += (size_t)NTOK * EE;

  const dim3 gC (CC  / 64, NTOK / 64);          // N = 768
  const dim3 gFF(FFD / 64, NTOK / 64);          // N = 3072
  const dim3 gV ((VV + 63) / 64, NTOK / 64);    // N = 95

  embed_kernel<<<(NTOK * CC) / 256, 256, 0, stream>>>(idx, tok_emb, pos_emb, x);

  for (int l = 0; l < LL; l++) {
    const size_t wofs = (size_t)l * CC * CC;
    // q,k,v projections (no bias in reference)
    gemm_bf16_wmma<0,0><<<gC, 128, 0, stream>>>(x, CC, Wq + wofs, CC, nullptr, q,    CC, NTOK, CC, CC, nullptr, 0, 0);
    gemm_bf16_wmma<0,0><<<gC, 128, 0, stream>>>(x, CC, Wk + wofs, CC, nullptr, kbuf, CC, NTOK, CC, CC, nullptr, 0, 0);
    gemm_bf16_wmma<0,0><<<gC, 128, 0, stream>>>(x, CC, Wv + wofs, CC, nullptr, vbuf, CC, NTOK, CC, CC, nullptr, 0, 0);
    // causal attention
    attn_kernel<<<BB * HH, 256, 0, stream>>>(q, kbuf, vbuf, obuf);
    // output projection
    gemm_bf16_wmma<0,0><<<gC, 128, 0, stream>>>(obuf, CC, Wo + wofs, CC, bo + (size_t)l * CC, y, CC, NTOK, CC, CC, nullptr, 0, 0);
    // x = LN(x + y)
    ln_kernel<true><<<NTOK, 256, 0, stream>>>(x, y, ln1_g + (size_t)l * CC, ln1_b + (size_t)l * CC, x);
    // gating -> normalized top-2 routing weights (0 for unrouted)
    gate_topk_kernel<<<NTOK, 32, 0, stream>>>(x, gate_W + (size_t)l * CC * EE, gate_b + (size_t)l * EE, route);
    // MoE: y = sum_e route[:,e] * (gelu(x@W1[e]+b1[e]) @ W2[e] + b2[e])
    zero_kernel<<<(NTOK * CC) / 256, 256, 0, stream>>>(y, NTOK * CC);
    for (int e = 0; e < EE; e++) {
      const float* W1e = W1 + ((size_t)l * EE + e) * CC * FFD;
      const float* b1e = b1 + ((size_t)l * EE + e) * FFD;
      const float* W2e = W2 + ((size_t)l * EE + e) * FFD * CC;
      const float* b2e = b2 + ((size_t)l * EE + e) * CC;
      gemm_bf16_wmma<1,0><<<gFF, 128, 0, stream>>>(x,    CC,  W1e, FFD, b1e, hbuf, FFD, NTOK, FFD, CC,  nullptr, 0, 0);
      gemm_bf16_wmma<0,1><<<gC,  128, 0, stream>>>(hbuf, FFD, W2e, CC,  b2e, y,    CC,  NTOK, CC,  FFD, route, EE, e);
    }
    // x = LN(x + y)
    ln_kernel<true><<<NTOK, 256, 0, stream>>>(x, y, ln2_g + (size_t)l * CC, ln2_b + (size_t)l * CC, x);
  }

  // final LN + LM head
  ln_kernel<false><<<NTOK, 256, 0, stream>>>(x, nullptr, lnf_g, lnf_b, x);
  gemm_bf16_wmma<0,0><<<gV, 128, 0, stream>>>(x, CC, head_W, VV, head_b, out, VV, NTOK, VV, CC, nullptr, 0, 0);
}
